// MPNN_40286793237043
// MI455X (gfx1250) — matrix operand order repaired and checked by reading, not compiled, not run on hardware
//
#include <hip/hip_runtime.h>

// Problem constants
#define BB 8
#define NN 64
#define F_IN 32
#define F_E 16
#define HH 64
#define MM 64
#define EH 50
#define TT 12
#define RD 200
#define NL 3

#define KTOT 3264            // EH*MM (3200) + 64 bias columns
#define NODES (BB*NN)        // 512
#define EDGES (BB*NN*NN)     // 32768

typedef _Float16 v16h __attribute__((ext_vector_type(16)));
typedef _Float16 v8h  __attribute__((ext_vector_type(8)));
typedef float    v8f  __attribute__((ext_vector_type(8)));

// ---------------- WMMA fragment loaders (CDNA5 16-bit layouts) ----------------
// A-matrix 16x32 f16: lanes 0-15 -> V0-3: K=0..7, V4-7: K=16..23
//                     lanes16-31 -> V0-3: K=8..15, V4-7: K=24..31
__device__ __forceinline__ v16h frag_a(const _Float16* rowbase, int lane) {
    const int hi = (lane >= 16) ? 1 : 0;
    v8h x0 = *(const v8h*)(rowbase + (hi ? 8  : 0));
    v8h x1 = *(const v8h*)(rowbase + (hi ? 24 : 16));
    v16h r;
#pragma unroll
    for (int i = 0; i < 8; ++i) { r[i] = x0[i]; r[i + 8] = x1[i]; }
    return r;
}
// B-matrix 32x16 f16 (col-major storage, K contiguous per column):
// lanes 0-15 hold K=0..7,16..23, lanes 16-31 hold K=8..15,24..31 (same order as A)
__device__ __forceinline__ v16h frag_b(const _Float16* colbase, int lane) {
    const int hi = (lane >= 16) ? 1 : 0;
    v8h x0 = *(const v8h*)(colbase + (hi ? 8  : 0));
    v8h x1 = *(const v8h*)(colbase + (hi ? 24 : 16));
    v16h r;
#pragma unroll
    for (int i = 0; i < 8; ++i) { r[i] = x0[i]; r[i + 8] = x1[i]; }
    return r;
}

// ---------------- Prep: h0 (f32 + f16) and node mask ----------------
__global__ void prep_h0_kernel(const float* __restrict__ h_in,
                               float* __restrict__ h0,
                               _Float16* __restrict__ h16,
                               float* __restrict__ mask) {
    int tid = blockIdx.x * blockDim.x + threadIdx.x;   // NODES*HH = 32768
    int node = tid >> 6, x = tid & 63;
    float v = (x < F_IN) ? h_in[node * F_IN + x] : 0.f;
    h0[tid] = v;
    h16[tid] = (_Float16)v;
    if (x == 0) {
        float s = 0.f;
        for (int f = 0; f < F_IN; ++f) s += h_in[node * F_IN + f];
        mask[node] = (s > 0.f) ? 1.f : 0.f;
    }
}

// ---------------- Prep: repacked We2^T (+be2 columns), col-major f16 ----------------
// Bt[c][h], c = kk*64+m for c<3200 -> We2[kk][m*64+h]; c in [3200,3264) -> be2[(c-3200)*64+h]
__global__ void prep_b_kernel(const float* __restrict__ We2,
                              const float* __restrict__ be2,
                              _Float16* __restrict__ Bt) {
    int tid = blockIdx.x * blockDim.x + threadIdx.x;   // KTOT*HH = 208896
    int c = tid >> 6, hh = tid & 63;
    float v;
    if (c < EH * MM) v = We2[(c >> 6) * (MM * HH) + (c & 63) * HH + hh];
    else             v = be2[(c - EH * MM) * HH + hh];
    Bt[tid] = (_Float16)v;
}

// ---------------- Prep: Rg[b][j][i*50+k] = g[b,i,j]*ReLU(e.We1+be1)[k]; [3200+i]=g ----------------
__global__ void prep_rg_kernel(const float* __restrict__ g,
                               const float* __restrict__ e,
                               const float* __restrict__ We1,
                               const float* __restrict__ be1,
                               _Float16* __restrict__ Rg) {
    int tid = blockIdx.x * blockDim.x + threadIdx.x;   // EDGES*64 = 2097152
    int t = tid & 63;
    int edge = tid >> 6;                  // b*4096 + i*64 + j
    int j = edge & 63, i = (edge >> 6) & 63, b = edge >> 12;
    float gv = g[b * (NN * NN) + i * NN + j];
    _Float16* row = Rg + (size_t)b * (NN * KTOT) + (size_t)j * KTOT;
    if (t < EH) {
        const float* ev = e + (size_t)edge * F_E;
        float a = be1[t];
#pragma unroll
        for (int f = 0; f < F_E; ++f) a += ev[f] * We1[f * EH + t];
        a = fmaxf(a, 0.f);
        row[i * EH + t] = (_Float16)(gv * a);
    } else if (t == EH) {
        row[EH * MM + i] = (_Float16)gv;
    }
}

// ---------------- U GEMM epilogue: scatter one 16x16 f32 tile into col-major U ----------------
// Branch on the tile index (wave-uniform): tiles 0..199 are weight columns, 200..203 bias cols.
__device__ __forceinline__ void u_store(int tnt, const v8f& acc, int tm, int lane,
                                        _Float16* __restrict__ U) {
    const int rbase = tm * 16 + ((lane < 16) ? 0 : 8);
    const int col = tnt * 16 + (lane & 15);
    if (tnt < 200) {                       // scalar branch: uniform across the wave
        const int kk = col >> 6, m = col & 63;
#pragma unroll
        for (int v = 0; v < 8; ++v) {
            int R = rbase + v;             // global row = b*64 + i
            int b_ = R >> 6, i = R & 63;
            U[b_ * (MM * KTOT) + m * KTOT + i * EH + kk] = (_Float16)acc[v];
        }
    } else {
        const int m = col - EH * MM;
#pragma unroll
        for (int v = 0; v < 8; ++v) {
            int R = rbase + v;
            int b_ = R >> 6, i = R & 63;
            U[b_ * (MM * KTOT) + m * KTOT + EH * MM + i] = (_Float16)acc[v];
        }
    }
}

// ---------------- U GEMM: (512x64 f16) @ (64x3264 f16, col-major) -> scattered f16 ----------------
// Each wave computes a 16x32 strip (2 N-tiles), reusing the A fragment.
__global__ __launch_bounds__(32) void u_gemm_kernel(const _Float16* __restrict__ h16,
                                                    const _Float16* __restrict__ Bt,
                                                    _Float16* __restrict__ U) {
    const int lane = threadIdx.x;
    const int tile = blockIdx.x;            // 32 * 102
    const int tm = tile / (KTOT / 32);
    const int tn2 = tile % (KTOT / 32);     // pair of N-tiles
    const int tn0 = tn2 * 2, tn1 = tn0 + 1;
    const _Float16* arow  = h16 + (tm * 16 + (lane & 15)) * HH;
    const _Float16* bcol0 = Bt + (tn0 * 16 + (lane & 15)) * HH;
    const _Float16* bcol1 = Bt + (tn1 * 16 + (lane & 15)) * HH;
    v8f c0 = {}, c1 = {};
#pragma unroll
    for (int ks = 0; ks < 2; ++ks) {
        v16h a  = frag_a(arow + ks * 32, lane);
        v16h b0 = frag_b(bcol0 + ks * 32, lane);
        v16h b1 = frag_b(bcol1 + ks * 32, lane);
        c0 = __builtin_amdgcn_wmma_f32_16x16x32_f16(false, a, false, b0, (short)0, c0, false, false);
        c1 = __builtin_amdgcn_wmma_f32_16x16x32_f16(false, a, false, b1, (short)0, c1, false, false);
    }
    u_store(tn0, c0, tm, lane, U);
    u_store(tn1, c1, tm, lane, U);
}

// ---------------- Message GEMM: per batch (64x3264) @ (3264x64) -> m (8,64,64) f32 ----------------
// Each wave computes a 2x2 block of 16x16 tiles (32x32 output), 4 WMMAs per 4 loads.
__global__ __launch_bounds__(32) void msg_gemm_kernel(const _Float16* __restrict__ Rg,
                                                      const _Float16* __restrict__ U,
                                                      float* __restrict__ mout) {
    const int lane = threadIdx.x;
    const int id = blockIdx.x;              // 8 * 2 * 2 = 32
    const int b = id >> 2;
    const int tm0 = ((id >> 1) & 1) * 2;    // row tiles tm0, tm0+1
    const int tn0 = (id & 1) * 2;           // col tiles tn0, tn0+1
    const _Float16* abase = Rg + (size_t)b * (NN * KTOT);
    const _Float16* bbase = U  + (size_t)b * (MM * KTOT);
    const _Float16* arow0 = abase + (size_t)(tm0 * 16 + (lane & 15)) * KTOT;
    const _Float16* arow1 = arow0 + (size_t)16 * KTOT;
    const _Float16* bcol0 = bbase + (size_t)(tn0 * 16 + (lane & 15)) * KTOT;
    const _Float16* bcol1 = bcol0 + (size_t)16 * KTOT;
    v8f c00 = {}, c01 = {}, c10 = {}, c11 = {};
    for (int ks = 0; ks < KTOT / 32; ++ks) {   // 102 K-steps
        v16h a0 = frag_a(arow0 + ks * 32, lane);
        v16h a1 = frag_a(arow1 + ks * 32, lane);
        v16h b0 = frag_b(bcol0 + ks * 32, lane);
        v16h b1 = frag_b(bcol1 + ks * 32, lane);
        c00 = __builtin_amdgcn_wmma_f32_16x16x32_f16(false, a0, false, b0, (short)0, c00, false, false);
        c01 = __builtin_amdgcn_wmma_f32_16x16x32_f16(false, a0, false, b1, (short)0, c01, false, false);
        c10 = __builtin_amdgcn_wmma_f32_16x16x32_f16(false, a1, false, b0, (short)0, c10, false, false);
        c11 = __builtin_amdgcn_wmma_f32_16x16x32_f16(false, a1, false, b1, (short)0, c11, false, false);
    }
    const int r0 = tm0 * 16 + ((lane < 16) ? 0 : 8);
    const int r1 = r0 + 16;
    const int col0 = tn0 * 16 + (lane & 15);
    const int col1 = col0 + 16;
    float* ob = mout + b * (NN * MM);
#pragma unroll
    for (int v = 0; v < 8; ++v) {
        ob[(r0 + v) * MM + col0] = c00[v];
        ob[(r0 + v) * MM + col1] = c01[v];
        ob[(r1 + v) * MM + col0] = c10[v];
        ob[(r1 + v) * MM + col1] = c11[v];
    }
}

// ---------------- Fused GRU cell (one block = one node) ----------------
__global__ __launch_bounds__(64) void gru_kernel(const float* __restrict__ m,
                                                 const float* __restrict__ hc,
                                                 const float* __restrict__ Wih,
                                                 const float* __restrict__ Whh,
                                                 const float* __restrict__ bih,
                                                 const float* __restrict__ bhh,
                                                 const float* __restrict__ mask,
                                                 float* __restrict__ hn,
                                                 _Float16* __restrict__ h16) {
    __shared__ float sm[HH], sh[HH];
    const int node = blockIdx.x;
    const int t = threadIdx.x;
    sm[t] = m[node * MM + t];
    sh[t] = hc[node * HH + t];
    __syncthreads();
    float gr = bih[t], gz = bih[HH + t], gn = bih[2 * HH + t];
    float hr = bhh[t], hz = bhh[HH + t], hnv = bhh[2 * HH + t];
    for (int x = 0; x < HH; ++x) {
        float mv = sm[x], hv = sh[x];
        gr += Wih[t * MM + x] * mv;
        gz += Wih[(HH + t) * MM + x] * mv;
        gn += Wih[(2 * HH + t) * MM + x] * mv;
        hr += Whh[t * HH + x] * hv;
        hz += Whh[(HH + t) * HH + x] * hv;
        hnv += Whh[(2 * HH + t) * HH + x] * hv;
    }
    float r = 1.f / (1.f + expf(-(gr + hr)));
    float z = 1.f / (1.f + expf(-(gz + hz)));
    float n = tanhf(gn + r * hnv);
    float out = ((1.f - z) * n + z * sh[t]) * mask[node];
    hn[node * HH + t] = out;
    h16[node * HH + t] = (_Float16)out;
}

// ---------------- Readout stage 1: i1 = ReLU([h,h0].Wi1+bi1), j1 = ReLU(h.Wj1+bj1) ----------------
__global__ void readout1_kernel(const float* __restrict__ h,
                                const float* __restrict__ h0,
                                const float* __restrict__ Wi1,
                                const float* __restrict__ bi1,
                                const float* __restrict__ Wj1,
                                const float* __restrict__ bj1,
                                float* __restrict__ i1,
                                float* __restrict__ j1) {
    int tid = blockIdx.x * blockDim.x + threadIdx.x;
    if (tid >= NODES * RD) return;
    int node = tid / RD, rd = tid % RD;
    const float* hv = h + node * HH;
    const float* h0v = h0 + node * HH;
    float ai = bi1[rd], aj = bj1[rd];
    for (int x = 0; x < HH; ++x) {
        ai += hv[x] * Wi1[x * RD + rd];
        aj += hv[x] * Wj1[x * RD + rd];
    }
    for (int x = 0; x < HH; ++x)
        ai += h0v[x] * Wi1[(HH + x) * RD + rd];
    i1[node * RD + rd] = fmaxf(ai, 0.f);
    j1[node * RD + rd] = fmaxf(aj, 0.f);
}

// ---------------- Readout stage 2: sum_v mask * sigmoid(i2) * j2 ----------------
__global__ __launch_bounds__(64) void readout2_kernel(const float* __restrict__ i1,
                                                      const float* __restrict__ j1,
                                                      const float* __restrict__ Wi2,
                                                      const float* __restrict__ bi2,
                                                      const float* __restrict__ Wj2,
                                                      const float* __restrict__ bj2,
                                                      const float* __restrict__ mask,
                                                      float* __restrict__ out) {
    __shared__ float acc[NN][TT];
    const int b = blockIdx.x, n = threadIdx.x;
    const int node = b * NN + n;
    const float* iv = i1 + node * RD;
    const float* jv = j1 + node * RD;
    const float mk = mask[node];
    for (int t = 0; t < TT; ++t) {
        float si = bi2[t], sj = bj2[t];
        for (int x = 0; x < RD; ++x) {
            si += iv[x] * Wi2[x * TT + t];
            sj += jv[x] * Wj2[x * TT + t];
        }
        float ia = 1.f / (1.f + expf(-si));
        acc[n][t] = mk * ia * sj;
    }
    __syncthreads();
    if (n < TT) {
        float s = 0.f;
        for (int x = 0; x < NN; ++x) s += acc[x][n];
        out[b * TT + n] = s;
    }
}

extern "C" void kernel_launch(void* const* d_in, const int* in_sizes, int n_in,
                              void* d_out, int out_size, void* d_ws, size_t ws_size,
                              hipStream_t stream) {
    const float* g    = (const float*)d_in[0];
    const float* h_in = (const float*)d_in[1];
    const float* e    = (const float*)d_in[2];
    const float* We1  = (const float*)d_in[3];
    const float* be1  = (const float*)d_in[4];
    const float* We2  = (const float*)d_in[5];
    const float* be2  = (const float*)d_in[6];
    const float* W_ih = (const float*)d_in[7];
    const float* W_hh = (const float*)d_in[8];
    const float* b_ih = (const float*)d_in[9];
    const float* b_hh = (const float*)d_in[10];
    const float* Wi1  = (const float*)d_in[11];
    const float* bi1  = (const float*)d_in[12];
    const float* Wi2  = (const float*)d_in[13];
    const float* bi2  = (const float*)d_in[14];
    const float* Wj1  = (const float*)d_in[15];
    const float* bj1  = (const float*)d_in[16];
    const float* Wj2  = (const float*)d_in[17];
    const float* bj2  = (const float*)d_in[18];
    float* out = (float*)d_out;

    char* ws = (char*)d_ws;
    size_t off = 0;
    auto alloc = [&](size_t bytes) {
        void* p = ws + off;
        off += (bytes + 255) & ~(size_t)255;
        return p;
    };
    float*    h0   = (float*)alloc(NODES * HH * sizeof(float));
    float*    hA   = (float*)alloc(NODES * HH * sizeof(float));
    float*    hB   = (float*)alloc(NODES * HH * sizeof(float));
    _Float16* h16  = (_Float16*)alloc(NODES * HH * sizeof(_Float16));
    float*    mask = (float*)alloc(NODES * sizeof(float));
    _Float16* Bt   = (_Float16*)alloc((size_t)KTOT * HH * sizeof(_Float16));
    _Float16* Rg   = (_Float16*)alloc((size_t)BB * NN * KTOT * sizeof(_Float16));
    _Float16* U    = (_Float16*)alloc((size_t)BB * MM * KTOT * sizeof(_Float16));
    float*    mb   = (float*)alloc(NODES * MM * sizeof(float));
    float*    i1   = (float*)alloc(NODES * RD * sizeof(float));
    float*    j1   = (float*)alloc(NODES * RD * sizeof(float));

    // One-time preprocessing (layer-invariant)
    prep_h0_kernel<<<(NODES * HH) / 256, 256, 0, stream>>>(h_in, h0, h16, mask);
    prep_b_kernel<<<(KTOT * HH) / 256, 256, 0, stream>>>(We2, be2, Bt);
    prep_rg_kernel<<<(EDGES * 64) / 256, 256, 0, stream>>>(g, e, We1, be1, Rg);

    // NL weight-tied message-passing layers
    float* hcur = h0;
    float* bufs[2] = {hA, hB};
    for (int layer = 0; layer < NL; ++layer) {
        u_gemm_kernel<<<(NODES / 16) * (KTOT / 32), 32, 0, stream>>>(h16, Bt, U);
        msg_gemm_kernel<<<BB * 2 * 2, 32, 0, stream>>>(Rg, U, mb);
        float* hnext = bufs[layer & 1];
        gru_kernel<<<NODES, HH, 0, stream>>>(mb, hcur, W_ih, W_hh, b_ih, b_hh,
                                             mask, hnext, h16);
        hcur = hnext;
    }

    // Readout
    readout1_kernel<<<(NODES * RD + 255) / 256, 256, 0, stream>>>(
        hcur, h0, Wi1, bi1, Wj1, bj1, i1, j1);
    readout2_kernel<<<BB, NN, 0, stream>>>(i1, j1, Wi2, bi2, Wj2, bj2, mask, out);
}
